// LinPolicy_28544352649408
// MI455X (gfx1250) — matrix operand order repaired and checked by reading, not compiled, not run on hardware
//
#include <hip/hip_runtime.h>
#include <math.h>

typedef __attribute__((ext_vector_type(16))) _Float16 v16h;
typedef __attribute__((ext_vector_type(8)))  _Float16 v8h;
typedef __attribute__((ext_vector_type(8)))  float    v8f;

// ---------------------------------------------------------------------------
// Weight prep: W [K,N] f32 row-major  ->  Wt [Npad, Kpad] f16 (each output
// column's K-run contiguous), zero padded. Bias padded to Npad f32.
// ---------------------------------------------------------------------------
__global__ __launch_bounds__(256) void prep_weights(
    const float* __restrict__ W, const float* __restrict__ b,
    _Float16* __restrict__ Wt, float* __restrict__ bp,
    int K, int N, int Kpad, int Npad)
{
    int i = blockIdx.x * blockDim.x + threadIdx.x;
    int tot = Npad * Kpad;
    if (i < tot) {
        int n = i / Kpad, k = i % Kpad;
        float v = (n < N && k < K) ? W[(size_t)k * N + n] : 0.0f;
        Wt[i] = (_Float16)v;
    }
    if (i < Npad) bp[i] = (i < N) ? b[i] : 0.0f;
}

// ---------------------------------------------------------------------------
// Position encodings (f16, K-padded)
// ---------------------------------------------------------------------------
__global__ __launch_bounds__(256) void encode_main(
    const float* __restrict__ agvs, _Float16* __restrict__ mainEnc, int B)
{
    int bIdx = blockIdx.x * blockDim.x + threadIdx.x;
    if (bIdx >= B) return;
    const float* x = agvs + ((size_t)bIdx * 201 + 1) * 16;
    _Float16* o = mainEnc + (size_t)bIdx * 32;
    float xv[16];
    #pragma unroll
    for (int i = 0; i < 16; i++) { xv[i] = x[i]; o[i] = (_Float16)xv[i]; }
    #pragma unroll
    for (int j = 0; j < 7; j++) {
        float c = xv[1 + 2 * j];
        o[16 + j] = (_Float16)sinf(c);
        o[23 + j] = (_Float16)cosf(c);
    }
    o[30] = (_Float16)0.0f; o[31] = (_Float16)0.0f;
}

__global__ __launch_bounds__(256) void encode_agv(
    const float* __restrict__ agvs, const _Float16* __restrict__ mainEnc,
    _Float16* __restrict__ dst, int b0, int rows)
{
    int r = blockIdx.x * blockDim.x + threadIdx.x;
    if (r >= rows) return;
    int bl = r / 200, rr = r % 200;
    int b = b0 + bl;
    const _Float16* m = mainEnc + (size_t)b * 32;
    _Float16* o = dst + (size_t)r * 64;
    #pragma unroll
    for (int i = 0; i < 30; i++) o[i] = m[i];
    const float* x = agvs + ((size_t)b * 201 + 1 + rr) * 16;
    float xv[16];
    #pragma unroll
    for (int i = 0; i < 16; i++) { xv[i] = x[i]; o[30 + i] = (_Float16)xv[i]; }
    #pragma unroll
    for (int j = 0; j < 7; j++) {
        float c = xv[1 + 2 * j];
        o[46 + j] = (_Float16)sinf(c);
        o[53 + j] = (_Float16)cosf(c);
    }
    #pragma unroll
    for (int i = 60; i < 64; i++) o[i] = (_Float16)0.0f;
}

__global__ __launch_bounds__(256) void encode_stat(
    const float* __restrict__ stat, const _Float16* __restrict__ mainEnc,
    _Float16* __restrict__ dst, int b0, int rows)
{
    int r = blockIdx.x * blockDim.x + threadIdx.x;
    if (r >= rows) return;
    int bl = r / 50, ss = r % 50;
    int b = b0 + bl;
    const _Float16* m = mainEnc + (size_t)b * 32;
    _Float16* o = dst + (size_t)r * 64;
    #pragma unroll
    for (int i = 0; i < 30; i++) o[i] = m[i];
    const float* x = stat + ((size_t)b * 50 + ss) * 16;
    float x0 = x[0];
    #pragma unroll
    for (int i = 0; i < 16; i++) o[30 + i] = (_Float16)x[i];
    o[46] = (_Float16)sinf(x0);
    o[47] = (_Float16)cosf(x0);
    #pragma unroll
    for (int i = 48; i < 64; i++) o[i] = (_Float16)0.0f;
}

// ---------------------------------------------------------------------------
// Fused GEMM + bias + optional ReLU via v_wmma_f32_16x16x32_f16.
//   A  : [M, Kpad] f16 row-major (row stride == Kpad)
//   Wt : [Npad, Kpad] f16 (output-column-major)
//   block = 256 threads = 8 waves; each wave computes a (16*MT)x(16*NT) tile
//   with MT*NT independent accumulator chains (register blocking for reuse).
//   grid = (Npad/(16*NT), ceil(M/(8*16*MT)))
// ---------------------------------------------------------------------------
template<int MT, int NT>
__global__ __launch_bounds__(256) void gemm_wmma(
    const _Float16* __restrict__ A, const _Float16* __restrict__ Wt,
    const float* __restrict__ bias, _Float16* __restrict__ Ch,
    float* __restrict__ Cf, int M, int Kpad, int ldC, int relu)
{
    int lane = threadIdx.x & 31;
    int wave = threadIdx.x >> 5;
    int tileM = (blockIdx.y * 8 + wave) * (16 * MT);
    if (tileM >= M) return;            // wave-uniform exit, EXEC stays all-1s
    int tileN = blockIdx.x * (16 * NT);
    int l15 = lane & 15;
    int g   = lane >> 4;

    const _Float16* aBase[MT];
    #pragma unroll
    for (int mt = 0; mt < MT; mt++) {
        int rowA = tileM + mt * 16 + l15;
        rowA = (rowA < M) ? rowA : (M - 1);      // branchless clamp (select)
        aBase[mt] = A + (size_t)rowA * Kpad + g * 8;
    }
    const _Float16* bBase[NT];
    #pragma unroll
    for (int nt = 0; nt < NT; nt++)
        bBase[nt] = Wt + (size_t)(tileN + nt * 16 + l15) * Kpad + g * 8;

    v8f acc[MT][NT];
    #pragma unroll
    for (int mt = 0; mt < MT; mt++)
        #pragma unroll
        for (int nt = 0; nt < NT; nt++)
            acc[mt][nt] = (v8f){};

    for (int kk = 0; kk < Kpad; kk += 32) {
        // A fragments: lane holds row M=lane&15; elems 0..7 = K kk+g*8..+7,
        // elems 8..15 = K kk+16+g*8..+7  (two b128 loads each)
        v16h av[MT];
        #pragma unroll
        for (int mt = 0; mt < MT; mt++) {
            v8h a0 = *(const v8h*)(aBase[mt] + kk);
            v8h a1 = *(const v8h*)(aBase[mt] + kk + 16);
            av[mt] = __builtin_shufflevector(a0, a1,
                0,1,2,3,4,5,6,7,8,9,10,11,12,13,14,15);
        }
        // B fragments: lane holds col N=lane&15; elems 0..7 = K kk+g*8..+7, elems 8..15 = K kk+16+g*8..+7
        v16h bv[NT];
        #pragma unroll
        for (int nt = 0; nt < NT; nt++) {
            v8h b0 = *(const v8h*)(bBase[nt] + kk);
            v8h b1 = *(const v8h*)(bBase[nt] + kk + 16);
            bv[nt] = __builtin_shufflevector(b0, b1,
                0,1,2,3,4,5,6,7,8,9,10,11,12,13,14,15);
        }
        #pragma unroll
        for (int mt = 0; mt < MT; mt++)
            #pragma unroll
            for (int nt = 0; nt < NT; nt++)
                acc[mt][nt] = __builtin_amdgcn_wmma_f32_16x16x32_f16(
                    false, av[mt], false, bv[nt], (short)0, acc[mt][nt],
                    false, false);
    }

    #pragma unroll
    for (int nt = 0; nt < NT; nt++) {
        int colB = tileN + nt * 16 + l15;
        float bb = bias[colB];
        #pragma unroll
        for (int mt = 0; mt < MT; mt++) {
            #pragma unroll
            for (int v = 0; v < 8; v++) {   // C layout: VGPR v -> M=v+8g, N=lane&15
                int row = tileM + mt * 16 + v + 8 * g;
                if (row >= M) continue;      // predication only after all WMMAs
                float val = acc[mt][nt][v] + bb;
                if (relu) val = fmaxf(val, 0.0f);
                if (Ch) Ch[(size_t)row * ldC + colB] = (_Float16)val;
                if (Cf) Cf[(size_t)row * ldC + colB] = val;
            }
        }
    }
}

// ---------------------------------------------------------------------------
// Max-pool over the set dimension: h [cb*R, 64] f16 -> features[b,colOff..+63]
// ---------------------------------------------------------------------------
__global__ __launch_bounds__(64) void maxpool64(
    const _Float16* __restrict__ h, _Float16* __restrict__ features,
    int R, int b0, int colOff)
{
    int bl = blockIdx.x;
    int j = threadIdx.x;
    const _Float16* p = h + (size_t)bl * R * 64 + j;
    float m = -3.0e38f;
    for (int r = 0; r < R; r++) m = fmaxf(m, (float)p[(size_t)r * 64]);
    features[(size_t)(b0 + bl) * 192 + colOff + j] = (_Float16)m;
}

// ---------------------------------------------------------------------------
// out[b, j] = (mask[b,j]-1)*1e8 + (j<9 ? logits[b,j] : 0)
// ---------------------------------------------------------------------------
__global__ __launch_bounds__(256) void finalize_out(
    const float* __restrict__ actOut, const float* __restrict__ mask,
    float* __restrict__ out, int total)
{
    int i = blockIdx.x * blockDim.x + threadIdx.x;
    if (i >= total) return;
    int b = i / 1809, j = i % 1809;
    float v = (mask[i] - 1.0f) * 100000000.0f;
    if (j < 9) v += actOut[(size_t)b * 16 + j];
    out[i] = v;
}

// ---------------------------------------------------------------------------
extern "C" void kernel_launch(void* const* d_in, const int* in_sizes, int n_in,
                              void* d_out, int out_size, void* d_ws, size_t ws_size,
                              hipStream_t stream) {
    (void)n_in; (void)out_size; (void)ws_size;
    const float* agvs = (const float*)d_in[0];
    const float* stat = (const float*)d_in[1];
    const float* mask = (const float*)d_in[2];
    float* out = (float*)d_out;
    int B = in_sizes[0] / (201 * 16);

    char* ws = (char*)d_ws;
    size_t off = 0;
    auto alloc = [&](size_t bytes) -> void* {
        off = (off + 255) & ~(size_t)255;
        void* p = ws + off;
        off += bytes;
        return p;
    };

    // layer table: {d_in index of W, K, N, Kpad, Npad}
    struct LD { int wIdx, K, N, Kpad, Npad; };
    const LD L[15] = {
        // main MLP  [30->128->128->128->64]
        {3, 30, 128, 32, 128}, {5, 128, 128, 128, 128},
        {7, 128, 128, 128, 128}, {9, 128, 64, 128, 64},
        // agv MLP   [60->256->256->128->64]
        {11, 60, 256, 64, 256}, {13, 256, 256, 256, 256},
        {15, 256, 128, 256, 128}, {17, 128, 64, 128, 64},
        // stat MLP  [48->256->256->128->64]
        {19, 48, 256, 64, 256}, {21, 256, 256, 256, 256},
        {23, 256, 128, 256, 128}, {25, 128, 64, 128, 64},
        // action MLP [192->256->64->9(pad16)]
        {27, 192, 256, 192, 256}, {29, 256, 64, 256, 64}, {31, 64, 9, 64, 16}
    };

    _Float16* Wt[15]; float* Bp[15];
    for (int i = 0; i < 15; i++) {
        Wt[i] = (_Float16*)alloc((size_t)L[i].Npad * L[i].Kpad * sizeof(_Float16));
        Bp[i] = (float*)alloc((size_t)L[i].Npad * sizeof(float));
        int tot = L[i].Npad * L[i].Kpad;
        prep_weights<<<(tot + 255) / 256, 256, 0, stream>>>(
            (const float*)d_in[L[i].wIdx], (const float*)d_in[L[i].wIdx + 1],
            Wt[i], Bp[i], L[i].K, L[i].N, L[i].Kpad, L[i].Npad);
    }

    int chunkB = (B < 128) ? B : 128;
    int maxRows = chunkB * 200;
    if (maxRows < B) maxRows = B;

    _Float16* mainEnc  = (_Float16*)alloc((size_t)B * 32 * sizeof(_Float16));
    _Float16* X        = (_Float16*)alloc((size_t)maxRows * 256 * sizeof(_Float16));
    _Float16* Y        = (_Float16*)alloc((size_t)maxRows * 256 * sizeof(_Float16));
    _Float16* encBuf   = (_Float16*)alloc((size_t)maxRows * 64 * sizeof(_Float16));
    _Float16* features = (_Float16*)alloc((size_t)B * 192 * sizeof(_Float16));
    float*    actOut   = (float*)alloc((size_t)B * 16 * sizeof(float));

    encode_main<<<(B + 255) / 256, 256, 0, stream>>>(agvs, mainEnc, B);

    auto gemm = [&](const _Float16* A, int li, _Float16* Ch, float* Cf,
                    int M, int ldC, int relu) {
        int Npad = L[li].Npad;
        if (Npad % 64 == 0) {
            // 32x64 macro-tile per wave: 8 accumulators, 12 loads / 8 WMMAs
            dim3 grid(Npad / 64, (M + 255) / 256);
            gemm_wmma<2, 4><<<grid, 256, 0, stream>>>(
                A, Wt[li], Bp[li], Ch, Cf, M, L[li].Kpad, ldC, relu);
        } else {
            dim3 grid(Npad / 16, (M + 127) / 128);
            gemm_wmma<1, 1><<<grid, 256, 0, stream>>>(
                A, Wt[li], Bp[li], Ch, Cf, M, L[li].Kpad, ldC, relu);
        }
    };

    // main agent path -> features[:, 0:64]
    gemm(mainEnc, 0, X, nullptr, B, 128, 1);
    gemm(X, 1, Y, nullptr, B, 128, 1);
    gemm(Y, 2, X, nullptr, B, 128, 1);
    gemm(X, 3, features, nullptr, B, 192, 1);

    for (int b0 = 0; b0 < B; b0 += chunkB) {
        int cb = (B - b0 < chunkB) ? (B - b0) : chunkB;

        int rowsA = cb * 200;
        encode_agv<<<(rowsA + 255) / 256, 256, 0, stream>>>(agvs, mainEnc, encBuf, b0, rowsA);
        gemm(encBuf, 4, X, nullptr, rowsA, 256, 1);
        gemm(X, 5, Y, nullptr, rowsA, 256, 1);
        gemm(Y, 6, X, nullptr, rowsA, 128, 1);
        gemm(X, 7, Y, nullptr, rowsA, 64, 1);
        maxpool64<<<cb, 64, 0, stream>>>(Y, features, 200, b0, 64);

        int rowsS = cb * 50;
        encode_stat<<<(rowsS + 255) / 256, 256, 0, stream>>>(stat, mainEnc, encBuf, b0, rowsS);
        gemm(encBuf, 8, X, nullptr, rowsS, 256, 1);
        gemm(X, 9, Y, nullptr, rowsS, 256, 1);
        gemm(Y, 10, X, nullptr, rowsS, 128, 1);
        gemm(X, 11, Y, nullptr, rowsS, 64, 1);
        maxpool64<<<cb, 64, 0, stream>>>(Y, features, 50, b0, 128);
    }

    // action head
    gemm(features, 12, X, nullptr, B, 256, 1);
    gemm(X, 13, Y, nullptr, B, 64, 1);
    gemm(Y, 14, nullptr, actOut, B, 16, 0);

    int total = B * 1809;
    finalize_out<<<(total + 255) / 256, 256, 0, stream>>>(actOut, mask, out, total);
}
